// ConvDiscriminator_6322191860134
// MI455X (gfx1250) — compile-verified, matrix operand order checked by reading, not run on hardware
//
#include <hip/hip_runtime.h>
#include <hip/hip_bf16.h>

// ---------------- types for WMMA ----------------
typedef __attribute__((ext_vector_type(16))) __bf16 bf16x16;
typedef __attribute__((ext_vector_type(8)))  float  floatx8;

union BF16Frag { bf16x16 v; uint4 q[2]; };

#define BB   4096      // batch
#define TT   34        // input time
#define DD   180       // input feature
#define TP   28        // T' = T-6
#define HH   64        // hidden
#define GG   192       // 3H
#define MROWS (TP*BB)  // 114688

// =======================================================================
// prep: zero BN stats, zero h0 slots, reorder conv1 weights, cvt GRU wts
// =======================================================================
__global__ __launch_bounds__(256)
void cd_prep_kernel(const float* __restrict__ c1w, float* __restrict__ wflat,
                    const float* __restrict__ wih, const float* __restrict__ whh,
                    __bf16* __restrict__ wihBf, __bf16* __restrict__ whhBf,
                    float* __restrict__ stats,
                    __bf16* __restrict__ h0a, __bf16* __restrict__ h0b)
{
    int i = blockIdx.x * blockDim.x + threadIdx.x;
    int n = gridDim.x * blockDim.x;
    for (int j = i; j < 48; j += n) stats[j] = 0.f;
    // wflat[co][k*180+ci] = conv1_w[co][ci][k]
    for (int j = i; j < 16 * 540; j += n) {
        int co = j / 540, r = j % 540, k = r / DD, ci = r % DD;
        wflat[j] = c1w[(co * DD + ci) * 3 + k];
    }
    for (int j = i; j < 3 * GG * HH; j += n) wihBf[j] = (__bf16)wih[j];
    for (int j = i; j < 4 * GG * HH; j += n) whhBf[j] = (__bf16)whh[j];
    for (int j = i; j < BB * HH; j += n) { h0a[j] = (__bf16)0.f; h0b[j] = (__bf16)0.f; }
}

// =======================================================================
// conv1: [B,34,180] -> [B,16,32] (+bias) and BN1 stats (sum, sumsq)
// one block per batch element
// =======================================================================
__global__ __launch_bounds__(128)
void cd_conv1_kernel(const float* __restrict__ poses,
                     const float* __restrict__ wflat,
                     const float* __restrict__ bias,
                     float* __restrict__ out1,
                     float* __restrict__ stats)
{
    __shared__ float sIn[TT * DD];      // 24.5 KB
    __shared__ float sWf[16 * 540];     // 34.6 KB
    __shared__ float sRed[128];
    int b = blockIdx.x, tid = threadIdx.x;
    for (int i = tid; i < TT * DD; i += 128) sIn[i] = poses[(long long)b * TT * DD + i];
    for (int i = tid; i < 16 * 540; i += 128) sWf[i] = wflat[i];
    __syncthreads();
    int co = tid >> 3;      // 16 channels x 8 threads
    int l0 = tid & 7;
    float s = 0.f, sq = 0.f;
    for (int j = 0; j < 4; ++j) {
        int l = l0 + 8 * j;                 // 0..31
        const float* xp = sIn + l * DD;     // contiguous 540 window
        const float* wp = sWf + co * 540;
        float acc = bias[co];
        #pragma unroll 4
        for (int k = 0; k < 540; ++k) acc += xp[k] * wp[k];
        out1[((long long)b * 16 + co) * 32 + l] = acc;
        s += acc; sq += acc * acc;
    }
    sRed[tid] = s; __syncthreads();
    if (l0 == 0) { float t = 0; for (int i = 0; i < 8; ++i) t += sRed[co * 8 + i]; atomicAdd(&stats[co], t); }
    __syncthreads();
    sRed[tid] = sq; __syncthreads();
    if (l0 == 0) { float t = 0; for (int i = 0; i < 8; ++i) t += sRed[co * 8 + i]; atomicAdd(&stats[16 + co], t); }
}

// =======================================================================
// BN finalize: scale/shift per channel
// =======================================================================
__global__ void cd_bnfin_kernel(const float* __restrict__ stats, int nc, float invN,
                                const float* __restrict__ g, const float* __restrict__ bb,
                                float* __restrict__ scsh)
{
    int c = threadIdx.x;
    if (c < nc) {
        float m = stats[c] * invN;
        float v = stats[nc + c] * invN - m * m;
        float sc = g[c] * rsqrtf(v + 1e-5f);
        scsh[c] = sc;
        scsh[nc + c] = bb[c] - m * sc;
    }
}

// =======================================================================
// conv2: BN1+leaky on the fly, conv 16->8 k3 -> out2 [B,8,30] + BN2 stats
// =======================================================================
__global__ __launch_bounds__(128)
void cd_conv2_kernel(const float* __restrict__ out1, const float* __restrict__ bn1,
                     const float* __restrict__ w2, const float* __restrict__ b2,
                     float* __restrict__ out2, float* __restrict__ stats2)
{
    __shared__ float sIn[16 * 32];
    __shared__ float sW[8 * 16 * 3];
    __shared__ float sRed[128];
    int b = blockIdx.x, tid = threadIdx.x;
    for (int i = tid; i < 16 * 32; i += 128) {
        int c = i >> 5;
        float v = out1[(long long)b * 512 + i] * bn1[c] + bn1[16 + c];
        sIn[i] = v >= 0.f ? v : 0.1f * v;
    }
    for (int i = tid; i < 384; i += 128) sW[i] = w2[i];
    __syncthreads();
    int co = tid >> 4;   // 8 channels x 16 threads
    int l0 = tid & 15;
    float s = 0.f, sq = 0.f;
    for (int j = 0; j < 2; ++j) {
        int l = l0 + 16 * j;
        if (l < 30) {
            float acc = b2[co];
            #pragma unroll
            for (int ci = 0; ci < 16; ++ci)
                #pragma unroll
                for (int k = 0; k < 3; ++k)
                    acc += sIn[ci * 32 + l + k] * sW[(co * 16 + ci) * 3 + k];
            out2[((long long)b * 8 + co) * 30 + l] = acc;
            s += acc; sq += acc * acc;
        }
    }
    sRed[tid] = s; __syncthreads();
    if (l0 == 0) { float t = 0; for (int i = 0; i < 16; ++i) t += sRed[co * 16 + i]; atomicAdd(&stats2[co], t); }
    __syncthreads();
    sRed[tid] = sq; __syncthreads();
    if (l0 == 0) { float t = 0; for (int i = 0; i < 16; ++i) t += sRed[co * 16 + i]; atomicAdd(&stats2[8 + co], t); }
}

// =======================================================================
// conv3: BN2+leaky, conv 8->8 k3 -> feat time-major [T'=28][B][8] f32
// =======================================================================
__global__ __launch_bounds__(128)
void cd_conv3_kernel(const float* __restrict__ out2, const float* __restrict__ bn2,
                     const float* __restrict__ w3, const float* __restrict__ b3,
                     float* __restrict__ feat)
{
    __shared__ float sIn[8 * 30];
    __shared__ float sW[8 * 8 * 3];
    int b = blockIdx.x, tid = threadIdx.x;
    for (int i = tid; i < 8 * 30; i += 128) {
        int c = i / 30;
        float v = out2[(long long)b * 240 + i] * bn2[c] + bn2[8 + c];
        sIn[i] = v >= 0.f ? v : 0.1f * v;
    }
    for (int i = tid; i < 192; i += 128) sW[i] = w3[i];
    __syncthreads();
    int co = tid >> 4;
    int l0 = tid & 15;
    for (int j = 0; j < 2; ++j) {
        int l = l0 + 16 * j;
        if (l < TP) {
            float acc = b3[co];
            #pragma unroll
            for (int ci = 0; ci < 8; ++ci)
                #pragma unroll
                for (int k = 0; k < 3; ++k)
                    acc += sIn[ci * 30 + l + k] * sW[(co * 8 + ci) * 3 + k];
            feat[((long long)l * BB + b) * 8 + co] = acc;
        }
    }
}

// =======================================================================
// xg layer0 (K=8, too thin for WMMA): xg[row][n] = b_ih[n] + feat[row]·W[n]
// =======================================================================
__global__ __launch_bounds__(256)
void cd_xg0_kernel(const float* __restrict__ feat, const float* __restrict__ w,
                   const float* __restrict__ bias, __bf16* __restrict__ xg)
{
    __shared__ float sW[GG * 8];
    __shared__ float sB[GG];
    int tid = threadIdx.x;
    for (int i = tid; i < GG * 8; i += 256) sW[i] = w[i];
    for (int i = tid; i < GG; i += 256) sB[i] = bias[i];
    __syncthreads();
    long long row = (long long)blockIdx.x * 256 + tid;
    if (row >= (long long)MROWS) return;
    float x[8];
    #pragma unroll
    for (int k = 0; k < 8; ++k) x[k] = feat[row * 8 + k];
    for (int n = 0; n < GG; ++n) {
        float acc = sB[n];
        #pragma unroll
        for (int k = 0; k < 8; ++k) acc += x[k] * sW[n * 8 + k];
        xg[row * GG + n] = (__bf16)acc;
    }
}

// =======================================================================
// xg layers 1..3: WMMA GEMM  XG[M,192] = X[M,64] * W^T + b  (bf16 in/out)
// block = 256 thr (8 waves), 128-row tile; each wave owns one 16-row
// subtile and sweeps all 12 column tiles. K=64 -> 2x wmma_f32_16x16x32_bf16.
// =======================================================================
__global__ __launch_bounds__(256)
void cd_xg_gemm_kernel(const __bf16* __restrict__ X, const __bf16* __restrict__ W,
                       const float* __restrict__ bias, __bf16* __restrict__ XG)
{
    __shared__ __align__(16) __bf16 sX[128 * HH];   // 16 KB
    __shared__ __align__(16) __bf16 sW[GG * HH];    // 24 KB
    __shared__ float sB[GG];
    int tid = threadIdx.x;
    long long base = (long long)blockIdx.x * 128;
    {
        const uint4* s = (const uint4*)(X + base * HH);
        uint4* d = (uint4*)sX;
        for (int i = tid; i < 128 * HH / 8; i += 256) d[i] = s[i];
    }
    {
        const uint4* s = (const uint4*)W;
        uint4* d = (uint4*)sW;
        for (int i = tid; i < GG * HH / 8; i += 256) d[i] = s[i];
    }
    for (int i = tid; i < GG; i += 256) sB[i] = bias[i];
    __syncthreads();
    int wave = tid >> 5, lane = tid & 31, half = lane >> 4, mn = lane & 15;
    for (int nt = 0; nt < 12; ++nt) {
        floatx8 acc = {};
        #pragma unroll
        for (int kk = 0; kk < 2; ++kk) {
            BF16Frag a, bm;
            // CDNA5 16-bit A layout: lane half h holds K = {8h..8h+7} U {16+8h..16+8h+7}
            const __bf16* ar = sX + (wave * 16 + mn) * HH + kk * 32;
            a.q[0] = *(const uint4*)(ar + half * 8);
            a.q[1] = *(const uint4*)(ar + 16 + half * 8);
            const __bf16* br = sW + (nt * 16 + mn) * HH + kk * 32;   // W row n = B column n
            bm.q[0] = *(const uint4*)(br + half * 8);
            bm.q[1] = *(const uint4*)(br + 16 + half * 8);
            acc = __builtin_amdgcn_wmma_f32_16x16x32_bf16(false, a.v, false, bm.v,
                                                          (short)0, acc, false, false);
        }
        #pragma unroll
        for (int r = 0; r < 8; ++r) {
            long long row = base + wave * 16 + r + 8 * half;  // C layout: VGPR r -> M=r (+8 for hi lanes)
            int col = nt * 16 + mn;
            XG[row * GG + col] = (__bf16)(acc[r] + sB[col]);
        }
    }
}

// =======================================================================
// GRU recurrent step (one per layer, per timestep):
//   hg = h_prev @ W_hh^T ; gates in-register; h_new written bf16
// block = 256 thr (8 waves) covers 64 batch rows. Each wave task computes
// the r/z/n tiles for the SAME (row-tile, hcol-tile) -> gate math needs no
// cross-wave exchange. 12 wmma per wave, 96 per block.
// =======================================================================
__global__ __launch_bounds__(256)
void cd_gru_step_kernel(const __bf16* __restrict__ hprev,  // [4096,64]
                        const __bf16* __restrict__ xg_t,   // [4096,192]
                        const __bf16* __restrict__ whh,    // [192,64] (n,k)
                        const float*  __restrict__ bhh,    // [192]
                        __bf16* __restrict__ hout)         // [4096,64]
{
    __shared__ __align__(16) __bf16 sW[GG * HH];   // 24 KB
    __shared__ __align__(16) __bf16 sH[64 * HH];   // 8 KB
    __shared__ float sBH[GG];
    int tid = threadIdx.x;
    int b0 = blockIdx.x * 64;
    {
        const uint4* s = (const uint4*)whh;
        uint4* d = (uint4*)sW;
        for (int i = tid; i < GG * HH / 8; i += 256) d[i] = s[i];
    }
    {
        const uint4* s = (const uint4*)(hprev + (long long)b0 * HH);
        uint4* d = (uint4*)sH;
        for (int i = tid; i < 64 * HH / 8; i += 256) d[i] = s[i];
    }
    for (int i = tid; i < GG; i += 256) sBH[i] = bhh[i];
    __syncthreads();
    int wave = tid >> 5, lane = tid & 31, half = lane >> 4, mn = lane & 15;
    #pragma unroll
    for (int task = 0; task < 2; ++task) {
        int tt = wave * 2 + task;        // 0..15
        int mt = tt >> 2;                // row tile 0..3
        int hct = tt & 3;                // hcol tile 0..3
        floatx8 aR = {}, aZ = {}, aN = {};
        #pragma unroll
        for (int kk = 0; kk < 2; ++kk) {
            BF16Frag a;
            const __bf16* ar = sH + (mt * 16 + mn) * HH + kk * 32;
            a.q[0] = *(const uint4*)(ar + half * 8);
            a.q[1] = *(const uint4*)(ar + 16 + half * 8);
            BF16Frag bR, bZ, bN;
            const __bf16* brR = sW + (0 * 64 + hct * 16 + mn) * HH + kk * 32;
            const __bf16* brZ = sW + (1 * 64 + hct * 16 + mn) * HH + kk * 32;
            const __bf16* brN = sW + (2 * 64 + hct * 16 + mn) * HH + kk * 32;
            bR.q[0] = *(const uint4*)(brR + half * 8); bR.q[1] = *(const uint4*)(brR + 16 + half * 8);
            bZ.q[0] = *(const uint4*)(brZ + half * 8); bZ.q[1] = *(const uint4*)(brZ + 16 + half * 8);
            bN.q[0] = *(const uint4*)(brN + half * 8); bN.q[1] = *(const uint4*)(brN + 16 + half * 8);
            aR = __builtin_amdgcn_wmma_f32_16x16x32_bf16(false, a.v, false, bR.v, (short)0, aR, false, false);
            aZ = __builtin_amdgcn_wmma_f32_16x16x32_bf16(false, a.v, false, bZ.v, (short)0, aZ, false, false);
            aN = __builtin_amdgcn_wmma_f32_16x16x32_bf16(false, a.v, false, bN.v, (short)0, aN, false, false);
        }
        int hc = hct * 16 + mn;
        #pragma unroll
        for (int r = 0; r < 8; ++r) {
            int m = mt * 16 + r + 8 * half;
            long long row = b0 + m;
            float ir = (float)xg_t[row * GG + hc];
            float iz = (float)xg_t[row * GG + 64 + hc];
            float in_ = (float)xg_t[row * GG + 128 + hc];
            float hr = aR[r] + sBH[hc];
            float hz = aZ[r] + sBH[64 + hc];
            float hn = aN[r] + sBH[128 + hc];
            float rg = 1.f / (1.f + __expf(-(ir + hr)));
            float zg = 1.f / (1.f + __expf(-(iz + hz)));
            float ng = tanhf(in_ + rg * hn);
            float hp = (float)sH[m * HH + hc];
            float hnew = ng - ng * zg + zg * hp;   // (1-z)*n + z*h
            hout[row * HH + hc] = (__bf16)hnew;
        }
    }
}

// =======================================================================
// head: y1[b,t] = out·h + b ; y[b] = sigmoid(y1[b,:]·out2_w + out2_b)
// =======================================================================
__global__ void cd_head1_kernel(const __bf16* __restrict__ hf, const float* __restrict__ ow,
                                const float* __restrict__ ob, float* __restrict__ y1)
{
    int idx = blockIdx.x * blockDim.x + threadIdx.x;
    if (idx >= BB * TP) return;
    int b = idx / TP, t = idx % TP;
    const __bf16* p = hf + ((long long)(t + 1) * BB + b) * HH;
    float acc = ob[0];
    #pragma unroll 8
    for (int h = 0; h < HH; ++h) acc += (float)p[h] * ow[h];
    y1[idx] = acc;
}

__global__ void cd_head2_kernel(const float* __restrict__ y1, const float* __restrict__ w2,
                                const float* __restrict__ b2, float* __restrict__ out)
{
    int b = blockIdx.x * blockDim.x + threadIdx.x;
    if (b >= BB) return;
    float acc = b2[0];
    #pragma unroll
    for (int t = 0; t < TP; ++t) acc += y1[b * TP + t] * w2[t];
    out[b] = 1.f / (1.f + __expf(-acc));
}

// =======================================================================
// host launcher
// =======================================================================
extern "C" void kernel_launch(void* const* d_in, const int* in_sizes, int n_in,
                              void* d_out, int out_size, void* d_ws, size_t ws_size,
                              hipStream_t stream)
{
    const float* poses    = (const float*)d_in[0];
    const float* conv1_w  = (const float*)d_in[1];
    const float* conv1_b  = (const float*)d_in[2];
    const float* bn1_g    = (const float*)d_in[3];
    const float* bn1_b    = (const float*)d_in[4];
    const float* conv2_w  = (const float*)d_in[5];
    const float* conv2_b  = (const float*)d_in[6];
    const float* bn2_g    = (const float*)d_in[7];
    const float* bn2_b    = (const float*)d_in[8];
    const float* conv3_w  = (const float*)d_in[9];
    const float* conv3_b  = (const float*)d_in[10];
    const float* gru0_wih = (const float*)d_in[11];
    const float* gru_wih  = (const float*)d_in[12];  // [3,192,64]
    const float* gru_whh  = (const float*)d_in[13];  // [4,192,64]
    const float* gru_bih  = (const float*)d_in[14];  // [4,192]
    const float* gru_bhh  = (const float*)d_in[15];  // [4,192]
    const float* out_w    = (const float*)d_in[16];
    const float* out_b    = (const float*)d_in[17];
    const float* out2_w   = (const float*)d_in[18];
    const float* out2_b   = (const float*)d_in[19];
    float* out = (float*)d_out;

    // ---- workspace carve (256B aligned) ----
    char* p = (char*)d_ws;
    auto alloc = [&](size_t bytes) -> void* {
        void* r = (void*)p;
        p += (bytes + 255) & ~(size_t)255;
        return r;
    };
    float*   wflat  = (float*)  alloc(16 * 540 * 4);
    __bf16*  wihBf  = (__bf16*) alloc((size_t)3 * GG * HH * 2);
    __bf16*  whhBf  = (__bf16*) alloc((size_t)4 * GG * HH * 2);
    float*   stats  = (float*)  alloc(48 * 4);             // [0..31] bn1, [32..47] bn2
    float*   bn1    = (float*)  alloc(32 * 4);
    float*   bn2    = (float*)  alloc(16 * 4);
    float*   out1   = (float*)  alloc((size_t)BB * 16 * 32 * 4);
    float*   out2b  = (float*)  alloc((size_t)BB * 8 * 30 * 4);
    float*   feat   = (float*)  alloc((size_t)TP * BB * 8 * 4);
    __bf16*  xg     = (__bf16*) alloc((size_t)TP * BB * GG * 2);
    __bf16*  bufA   = (__bf16*) alloc((size_t)(TP + 1) * BB * HH * 2);
    __bf16*  bufB   = (__bf16*) alloc((size_t)(TP + 1) * BB * HH * 2);
    float*   y1     = (float*)  alloc((size_t)BB * TP * 4);

    // ---- prep ----
    cd_prep_kernel<<<256, 256, 0, stream>>>(conv1_w, wflat, gru_wih, gru_whh,
                                            wihBf, whhBf, stats, bufA, bufB);

    // ---- conv / BN pipeline ----
    cd_conv1_kernel<<<BB, 128, 0, stream>>>(poses, wflat, conv1_b, out1, stats);
    cd_bnfin_kernel<<<1, 32, 0, stream>>>(stats, 16, 1.f / (float)(BB * 32), bn1_g, bn1_b, bn1);
    cd_conv2_kernel<<<BB, 128, 0, stream>>>(out1, bn1, conv2_w, conv2_b, out2b, stats + 32);
    cd_bnfin_kernel<<<1, 32, 0, stream>>>(stats + 32, 8, 1.f / (float)(BB * 30), bn2_g, bn2_b, bn2);
    cd_conv3_kernel<<<BB, 128, 0, stream>>>(out2b, bn2, conv3_w, conv3_b, feat);

    // ---- 4 stacked GRU layers ----
    const __bf16* prevOut = nullptr;
    for (int layer = 0; layer < 4; ++layer) {
        if (layer == 0) {
            cd_xg0_kernel<<<MROWS / 256, 256, 0, stream>>>(feat, gru0_wih,
                                                           gru_bih + 0, xg);
        } else {
            cd_xg_gemm_kernel<<<MROWS / 128, 256, 0, stream>>>(
                prevOut, wihBf + (size_t)(layer - 1) * GG * HH,
                gru_bih + (size_t)layer * GG, xg);
        }
        __bf16* obuf = (layer & 1) ? bufB : bufA;
        const __bf16* whh_l = whhBf + (size_t)layer * GG * HH;
        const float* bhh_l = gru_bhh + (size_t)layer * GG;
        for (int t = 0; t < TP; ++t) {
            cd_gru_step_kernel<<<BB / 64, 256, 0, stream>>>(
                obuf + (size_t)t * BB * HH,          // h_{t-1} (slot 0 zeroed)
                xg + (size_t)t * BB * GG,
                whh_l, bhh_l,
                obuf + (size_t)(t + 1) * BB * HH);   // h_t
        }
        prevOut = obuf + (size_t)BB * HH;            // slots 1..28
    }

    // ---- head ----
    cd_head1_kernel<<<(BB * TP) / 256, 256, 0, stream>>>(bufB, out_w, out_b, y1);
    cd_head2_kernel<<<BB / 256, 256, 0, stream>>>(y1, out2_w, out2_b, out);
}
